// Compressor_47476568490001
// MI455X (gfx1250) — matrix operand order repaired and checked by reading, not compiled, not run on hardware
//
#include <hip/hip_runtime.h>
#include <hip/hip_bf16.h>
#include <math.h>

// ---------------------------------------------------------------------------
// CDNA5 (gfx1250) wave32 WMMA types
// ---------------------------------------------------------------------------
typedef __attribute__((ext_vector_type(16))) _Float16 v16h;
typedef __attribute__((ext_vector_type(8)))  float    v8f;

#define NWAVES 4
#define BLOCK  (NWAVES * 32)

// ---------------------------------------------------------------------------
// WMMA fragment loaders (layouts per CDNA5 ISA 7.12.2)
// 16-bit A 16x32: lanes 0-15 row M=lane; VGPR i<4 -> K=2i(+8 hi half),
//                 VGPR i>=4 -> K=16+2(i-4)(+8 hi half)
// 16-bit B 32x16: lane%16 = column N; same K order as A: lanes<16 K=0..7,16..23; lanes>=16 K=8..15,24..31
// f32 C/D 16x16 : lane%16 = column N; VGPR j -> M = j + 8*(lane/16)
// All f16 operand buffers are zero-padded so loads are unconditional and
// contiguous (vectorizable to b64/b128) -- no EXEC-masked scalar loads.
// ---------------------------------------------------------------------------
__device__ inline v16h wmma_a_from_lds(const _Float16* __restrict__ buf, int stride,
                                       int m_base, int k_base, int lane) {
  int lo = lane & 15, hi = (lane >> 4) & 1;
  const _Float16* row = buf + (size_t)(m_base + lo) * stride + k_base + hi * 8;
  v16h a;
#pragma unroll
  for (int i = 0; i < 4; ++i) { a[2 * i] = row[2 * i]; a[2 * i + 1] = row[2 * i + 1]; }
#pragma unroll
  for (int i = 4; i < 8; ++i) { a[2 * i] = row[16 + 2 * (i - 4)]; a[2 * i + 1] = row[16 + 2 * (i - 4) + 1]; }
  return a;
}

__device__ inline v16h wmma_a_from_gh(const _Float16* __restrict__ A, int lda,
                                      int m_base, int k_base, int lane) {
  int lo = lane & 15, hi = (lane >> 4) & 1;
  const _Float16* row = A + (size_t)(m_base + lo) * lda + k_base + hi * 8;
  row = (const _Float16*)__builtin_assume_aligned(row, 16);
  v16h a;
#pragma unroll
  for (int i = 0; i < 4; ++i) { a[2 * i] = row[2 * i]; a[2 * i + 1] = row[2 * i + 1]; }
#pragma unroll
  for (int i = 4; i < 8; ++i) { a[2 * i] = row[16 + 2 * (i - 4)]; a[2 * i + 1] = row[16 + 2 * (i - 4) + 1]; }
  return a;
}

// Transposed A: element (m,k) at A[k*lda + m]  (used for att^T)
__device__ inline v16h wmma_at_from_gh(const _Float16* __restrict__ A, int lda,
                                       int m_base, int k_base, int lane) {
  int lo = lane & 15, hi = (lane >> 4) & 1;
  int m = m_base + lo;
  v16h a;
#pragma unroll
  for (int i = 0; i < 8; ++i) {
    int k = k_base + 2 * (i & 3) + ((i >= 4) ? 16 : 0) + (hi ? 8 : 0);
    a[2 * i]     = A[(size_t)k * lda + m];
    a[2 * i + 1] = A[(size_t)(k + 1) * lda + m];
  }
  return a;
}

__device__ inline v16h wmma_b_from_gh(const _Float16* __restrict__ Bm, int ldb,
                                      int n_base, int k_base, int lane) {
  int lo = lane & 15, hi = (lane >> 4) & 1;
  const _Float16* row = Bm + (size_t)(n_base + lo) * ldb + k_base + hi * 8;
  row = (const _Float16*)__builtin_assume_aligned(row, 16);
  v16h b;
#pragma unroll
  for (int i = 0; i < 8; ++i) { int o = 2 * i + ((i >= 4) ? 8 : 0); b[2 * i] = row[o]; b[2 * i + 1] = row[o + 1]; }
  return b;
}

// Transposed B: element (n,k) at Bm[k*ldb + n]  (used for v as [i,c])
__device__ inline v16h wmma_bt_from_gh(const _Float16* __restrict__ Bm, int ldb,
                                       int n_base, int k_base, int lane) {
  int lo = lane & 15, hi = (lane >> 4) & 1;
  int n = n_base + lo;
  v16h b;
#pragma unroll
  for (int i = 0; i < 8; ++i) {
    int k = k_base + hi * 8 + 2 * (i & 3) + ((i >= 4) ? 16 : 0);
    b[2 * i]     = Bm[(size_t)k * ldb + n];
    b[2 * i + 1] = Bm[(size_t)(k + 1) * ldb + n];
  }
  return b;
}

// ---------------------------------------------------------------------------
// f32 -> zero-padded f16 conversion (weights / padded activations), done once
// ---------------------------------------------------------------------------
__global__ void cvt_pad_kernel(const float* __restrict__ src, _Float16* __restrict__ dst,
                               int rows, int kact, int kpad) {
  int t = blockIdx.x * blockDim.x + threadIdx.x;
  if (t >= rows * kpad) return;
  int r = t / kpad, k = t % kpad;
  float v = (k < kact) ? src[(size_t)r * kact + k] : 0.f;
  dst[t] = (_Float16)v;
}

// ---------------------------------------------------------------------------
// Farthest point sampling: one block per batch
// ---------------------------------------------------------------------------
__global__ __launch_bounds__(256)
void fps_kernel(const float* __restrict__ xyz, int N, int npoint,
                float* __restrict__ dist, int* __restrict__ fidx,
                float* __restrict__ new_xyz) {
  int b = blockIdx.x, tid = threadIdx.x;
  __shared__ float smax[256];
  __shared__ int   simax[256];
  __shared__ int   cur;
  float* db = dist + (size_t)b * N;
  for (int i = tid; i < N; i += 256) db[i] = 1e10f;
  if (tid == 0) cur = 0;
  __syncthreads();
  for (int it = 0; it < npoint; ++it) {
    int c = cur;
    float cx = xyz[((size_t)b * N + c) * 3 + 0];
    float cy = xyz[((size_t)b * N + c) * 3 + 1];
    float cz = xyz[((size_t)b * N + c) * 3 + 2];
    if (tid == 0) {
      fidx[(size_t)b * npoint + it] = c;
      new_xyz[((size_t)b * npoint + it) * 3 + 0] = cx;
      new_xyz[((size_t)b * npoint + it) * 3 + 1] = cy;
      new_xyz[((size_t)b * npoint + it) * 3 + 2] = cz;
    }
    float best = -1.f; int bi = 0;
    for (int i = tid; i < N; i += 256) {
      float dx = xyz[((size_t)b * N + i) * 3 + 0] - cx;
      float dy = xyz[((size_t)b * N + i) * 3 + 1] - cy;
      float dz = xyz[((size_t)b * N + i) * 3 + 2] - cz;
      float d2 = dx * dx + dy * dy + dz * dz;
      float d  = fminf(db[i], d2);
      db[i] = d;
      if (d > best) { best = d; bi = i; }
    }
    smax[tid] = best; simax[tid] = bi;
    __syncthreads();
    for (int o = 128; o > 0; o >>= 1) {
      if (tid < o) {
        if (smax[tid + o] > smax[tid]) { smax[tid] = smax[tid + o]; simax[tid] = simax[tid + o]; }
      }
      __syncthreads();
    }
    if (tid == 0) cur = simax[0];
    __syncthreads();
  }
}

// ---------------------------------------------------------------------------
// Ball query: one thread per centroid, first-K-in-ball, pad with first hit
// ---------------------------------------------------------------------------
__global__ void ball_query_kernel(const float* __restrict__ xyz, int N,
                                  const float* __restrict__ new_xyz, int S, int total,
                                  float r2, int K, int* __restrict__ gidx) {
  int t = blockIdx.x * blockDim.x + threadIdx.x;
  if (t >= total) return;
  int b = t / S;
  const float* q = new_xyz + (size_t)t * 3;
  float qx = q[0], qy = q[1], qz = q[2];
  const float* p = xyz + (size_t)b * N * 3;
  int* out = gidx + (size_t)t * K;
  int cnt = 0, first = 0; bool has = false;
  for (int i = 0; i < N && cnt < K; ++i) {
    __builtin_prefetch(p + (i + 16) * 3, 0, 1);
    float dx = p[i * 3 + 0] - qx;
    float dy = p[i * 3 + 1] - qy;
    float dz = p[i * 3 + 2] - qz;
    float d2 = dx * dx + dy * dy + dz * dz;
    if (d2 <= r2) {
      if (!has) { has = true; first = i; }
      out[cnt++] = i;
    }
  }
  for (int j = cnt; j < K; ++j) out[j] = has ? first : 0;
}

// ---------------------------------------------------------------------------
// One WMMA MLP layer over an LDS tile: out = relu((in @ Wh^T) * g + b)
// Wh is pre-padded f16 [cout, kpad].  Each wave computes a 16x32 output tile
// (two accumulators sharing one A fragment -> back-to-back WMMAs fill the
// WMMA->VALU hazard window and halve A traffic).
// ---------------------------------------------------------------------------
template <int ROWS>
__device__ inline void sa_layer(const _Float16* __restrict__ in, int in_stride, int kpad,
                                _Float16* __restrict__ out, int out_stride, int cout,
                                const _Float16* __restrict__ Wh,
                                const float* __restrict__ g,
                                const float* __restrict__ bb, int tid) {
  int wave = tid >> 5, lane = tid & 31;
  const int MT = ROWS / 16;
  int NP = cout / 32, KT = kpad / 32;
  for (int t = wave; t < MT * NP; t += NWAVES) {
    int mt = t / NP, np = (t % NP) * 2;
    v8f acc0 = {}, acc1 = {};
    for (int kc = 0; kc < KT; ++kc) {
      v16h a  = wmma_a_from_lds(in, in_stride, mt * 16, kc * 32, lane);
      v16h b0 = wmma_b_from_gh(Wh, kpad, np * 16, kc * 32, lane);
      v16h b1 = wmma_b_from_gh(Wh, kpad, (np + 1) * 16, kc * 32, lane);
      acc0 = __builtin_amdgcn_wmma_f32_16x16x32_f16(false, a, false, b0, (short)0, acc0, false, false);
      acc1 = __builtin_amdgcn_wmma_f32_16x16x32_f16(false, a, false, b1, (short)0, acc1, false, false);
    }
    int lo = lane & 15, hi = (lane >> 4) & 1;
    int col0 = np * 16 + lo, col1 = col0 + 16;
    float g0 = g[col0], bv0 = bb[col0];
    float g1 = g[col1], bv1 = bb[col1];
#pragma unroll
    for (int j = 0; j < 8; ++j) {
      int m = mt * 16 + j + hi * 8;
      float v0 = acc0[j] * g0 + bv0;
      float v1 = acc1[j] * g1 + bv1;
      out[(size_t)m * out_stride + col0] = (_Float16)(v0 > 0.f ? v0 : 0.f);
      out[(size_t)m * out_stride + col1] = (_Float16)(v1 > 0.f ? v1 : 0.f);
    }
  }
}

// Last layer with fused max-over-rows; partial[mt*cout + col]
template <int ROWS>
__device__ inline void sa_layer_max(const _Float16* __restrict__ in, int in_stride, int kpad,
                                    const _Float16* __restrict__ Wh,
                                    const float* __restrict__ g,
                                    const float* __restrict__ bb, int cout,
                                    float* __restrict__ partial, int tid) {
  int wave = tid >> 5, lane = tid & 31;
  const int MT = ROWS / 16;
  int NP = cout / 32, KT = kpad / 32;
  for (int t = wave; t < MT * NP; t += NWAVES) {
    int mt = t / NP, np = (t % NP) * 2;
    v8f acc0 = {}, acc1 = {};
    for (int kc = 0; kc < KT; ++kc) {
      v16h a  = wmma_a_from_lds(in, in_stride, mt * 16, kc * 32, lane);
      v16h b0 = wmma_b_from_gh(Wh, kpad, np * 16, kc * 32, lane);
      v16h b1 = wmma_b_from_gh(Wh, kpad, (np + 1) * 16, kc * 32, lane);
      acc0 = __builtin_amdgcn_wmma_f32_16x16x32_f16(false, a, false, b0, (short)0, acc0, false, false);
      acc1 = __builtin_amdgcn_wmma_f32_16x16x32_f16(false, a, false, b1, (short)0, acc1, false, false);
    }
    int lo = lane & 15, hi = (lane >> 4) & 1;
    int col0 = np * 16 + lo, col1 = col0 + 16;
    float g0 = g[col0], bv0 = bb[col0];
    float g1 = g[col1], bv1 = bb[col1];
    float m0 = -3.0e38f, m1 = -3.0e38f;
#pragma unroll
    for (int j = 0; j < 8; ++j) {
      float v0 = acc0[j] * g0 + bv0; v0 = v0 > 0.f ? v0 : 0.f;
      float v1 = acc1[j] * g1 + bv1; v1 = v1 > 0.f ? v1 : 0.f;
      m0 = fmaxf(m0, v0);
      m1 = fmaxf(m1, v1);
    }
    m0 = fmaxf(m0, __shfl_xor(m0, 16, 32));   // rows 0-7 vs rows 8-15
    m1 = fmaxf(m1, __shfl_xor(m1, 16, 32));
    if (hi == 0) {
      partial[mt * cout + col0] = m0;
      partial[mt * cout + col1] = m1;
    }
  }
}

// ---------------------------------------------------------------------------
// Set-abstraction kernel: one block per (batch, centroid) group
// ---------------------------------------------------------------------------
template <int ROWS, int CIN, int CIN_PAD, int C1, int C2, int C3, int CA, int CB, typename OutT>
__global__ __launch_bounds__(BLOCK)
void sa_mlp_kernel(const float* __restrict__ xyz, int N,
                   const float* __restrict__ feats, int FDIM,
                   const float* __restrict__ new_xyz,
                   const int* __restrict__ gidx, int S,
                   const _Float16* __restrict__ w1, const float* __restrict__ g1, const float* __restrict__ b1,
                   const _Float16* __restrict__ w2, const float* __restrict__ g2, const float* __restrict__ b2,
                   const _Float16* __restrict__ w3, const float* __restrict__ g3, const float* __restrict__ b3,
                   OutT* __restrict__ outf) {
  __shared__ _Float16 bufA[ROWS * CA];
  __shared__ _Float16 bufB[ROWS * CB];
  __shared__ float partial[(ROWS / 16) * C3];
  int b = blockIdx.x / S, s = blockIdx.x % S;
  int tid = threadIdx.x;
  const int* gi = gidx + ((size_t)b * S + s) * ROWS;
  const float* cz = new_xyz + ((size_t)b * S + s) * 3;
  for (int e = tid; e < ROWS * CIN_PAD; e += BLOCK) {
    int r = e / CIN_PAD, c = e % CIN_PAD;
    float v = 0.f;
    int pi = gi[r];
    if (c < 3)        v = xyz[((size_t)b * N + pi) * 3 + c] - cz[c];
    else if (c < CIN) v = feats[((size_t)b * N + pi) * FDIM + (c - 3)];
    bufA[r * CA + c] = (_Float16)v;
  }
  __syncthreads();
  sa_layer<ROWS>(bufA, CA, CIN_PAD, bufB, CB, C1, w1, g1, b1, tid);
  __syncthreads();
  sa_layer<ROWS>(bufB, CB, C1, bufA, CA, C2, w2, g2, b2, tid);
  __syncthreads();
  sa_layer_max<ROWS>(bufA, CA, C2, w3, g3, b3, C3, partial, tid);
  __syncthreads();
  for (int c = tid; c < C3; c += BLOCK) {
    float m = partial[c];
#pragma unroll
    for (int mt = 1; mt < ROWS / 16; ++mt) m = fmaxf(m, partial[mt * C3 + c]);
    outf[((size_t)b * S + s) * C3 + c] = (OutT)m;
  }
}

// ---------------------------------------------------------------------------
// Generic batched f16 WMMA GEMM: C = A @ B^T (+bias)(+resid)
// Each wave computes a 16x32 tile (two accumulators, shared A fragment).
// batch = blockIdx.y
// ---------------------------------------------------------------------------
template <bool TA, bool TB>
__global__ __launch_bounds__(BLOCK)
void gemm_h_kernel(const _Float16* __restrict__ A, int lda, long sA,
                   const _Float16* __restrict__ Bm, int ldb, long sB,
                   _Float16* __restrict__ C, int ldc, long sC,
                   int M, int Nn, int kpad,
                   const float* __restrict__ bias,
                   const _Float16* __restrict__ resid, long sR) {
  int batch = blockIdx.y;
  const _Float16* Ab = A + (size_t)batch * sA;
  const _Float16* Bb = Bm + (size_t)batch * sB;
  _Float16* Cb = C + (size_t)batch * sC;
  const _Float16* Rb = resid ? resid + (size_t)batch * sR : nullptr;
  int wave = threadIdx.x >> 5, lane = threadIdx.x & 31;
  int NP = Nn / 32;
  int tiles = (M / 16) * NP;
  int t = blockIdx.x * NWAVES + wave;
  if (t < tiles) {
    int mt = t / NP, np = (t % NP) * 2;
    v8f acc0 = {}, acc1 = {};
    for (int kc = 0; kc < kpad / 32; ++kc) {
      v16h a = TA ? wmma_at_from_gh(Ab, lda, mt * 16, kc * 32, lane)
                  : wmma_a_from_gh(Ab, lda, mt * 16, kc * 32, lane);
      v16h b0 = TB ? wmma_bt_from_gh(Bb, ldb, np * 16, kc * 32, lane)
                   : wmma_b_from_gh(Bb, ldb, np * 16, kc * 32, lane);
      v16h b1 = TB ? wmma_bt_from_gh(Bb, ldb, (np + 1) * 16, kc * 32, lane)
                   : wmma_b_from_gh(Bb, ldb, (np + 1) * 16, kc * 32, lane);
      acc0 = __builtin_amdgcn_wmma_f32_16x16x32_f16(false, a, false, b0, (short)0, acc0, false, false);
      acc1 = __builtin_amdgcn_wmma_f32_16x16x32_f16(false, a, false, b1, (short)0, acc1, false, false);
    }
    int lo = lane & 15, hi = (lane >> 4) & 1;
    int col0 = np * 16 + lo, col1 = col0 + 16;
    float bv0 = bias ? bias[col0] : 0.f;
    float bv1 = bias ? bias[col1] : 0.f;
#pragma unroll
    for (int j = 0; j < 8; ++j) {
      int m = mt * 16 + j + hi * 8;
      float v0 = acc0[j] + bv0;
      float v1 = acc1[j] + bv1;
      if (Rb) {
        v0 += (float)Rb[(size_t)m * ldc + col0];
        v1 += (float)Rb[(size_t)m * ldc + col1];
      }
      Cb[(size_t)m * ldc + col0] = (_Float16)v0;
      Cb[(size_t)m * ldc + col1] = (_Float16)v1;
    }
  }
}

// ---------------------------------------------------------------------------
// Row softmax then column-sum normalization, in place on f16 att
// ---------------------------------------------------------------------------
__global__ void attn_softmax_kernel(_Float16* __restrict__ att, int S) {
  int b = blockIdx.x;
  _Float16* ab = att + (size_t)b * S * S;
  int i = threadIdx.x;
  float mx = -3.0e38f;
  for (int j = 0; j < S; ++j) mx = fmaxf(mx, (float)ab[i * S + j]);
  float sum = 0.f;
  for (int j = 0; j < S; ++j) { float e = expf((float)ab[i * S + j] - mx); ab[i * S + j] = (_Float16)e; sum += e; }
  float inv = 1.f / sum;
  for (int j = 0; j < S; ++j) ab[i * S + j] = (_Float16)((float)ab[i * S + j] * inv);
  __syncthreads();
  int j = i;  // thread i owns column j
  float cs = 0.f;
  for (int r = 0; r < S; ++r) cs += (float)ab[r * S + j];
  float cinv = 1.f / (1e-9f + cs);
  for (int r = 0; r < S; ++r) ab[r * S + j] = (_Float16)((float)ab[r * S + j] * cinv);
}

__global__ void sub_h_kernel(const _Float16* __restrict__ a, const _Float16* __restrict__ b,
                             _Float16* __restrict__ c, int n) {
  int i = blockIdx.x * blockDim.x + threadIdx.x;
  if (i < n) c[i] = (_Float16)((float)a[i] - (float)b[i]);
}

// out[b,c] = max_i( x[b,i,c] + relu(d[b,i,c]*gamma[c]+beta[c]) ) + noise[b,c] - 0.5
__global__ void final_kernel(const _Float16* __restrict__ x, const _Float16* __restrict__ d,
                             const float* __restrict__ gamma, const float* __restrict__ beta,
                             const float* __restrict__ noise, float* __restrict__ out,
                             int S, int C) {
  int t = blockIdx.x * blockDim.x + threadIdx.x;
  int b = t / C, c = t % C;
  float g = gamma[c], be = beta[c];
  float m = -3.0e38f;
  for (int i = 0; i < S; ++i) {
    float dv = (float)d[((size_t)b * S + i) * C + c] * g + be;
    dv = dv > 0.f ? dv : 0.f;
    float y = (float)x[((size_t)b * S + i) * C + c] + dv;
    m = fmaxf(m, y);
  }
  out[t] = m + noise[t] - 0.5f;
}

// ---------------------------------------------------------------------------
// Host launch
// ---------------------------------------------------------------------------
static inline size_t align256(size_t x) { return (x + 255) & ~(size_t)255; }

extern "C" void kernel_launch(void* const* d_in, const int* in_sizes, int n_in,
                              void* d_out, int out_size, void* d_ws, size_t ws_size,
                              hipStream_t stream) {
  (void)in_sizes; (void)n_in; (void)out_size; (void)ws_size;
  const int B = 32, N = 8192;
  const int S1 = 512, K1 = 32;
  const int S2 = 128, K2 = 64;
  const int C1F = 128, C2F = 256;

  const float* xyz   = (const float*)d_in[0];
  const float* noise = (const float*)d_in[1];
  const float *w1a = (const float*)d_in[2],  *g1a = (const float*)d_in[3],  *b1a = (const float*)d_in[4];
  const float *w1b = (const float*)d_in[5],  *g1b = (const float*)d_in[6],  *b1b = (const float*)d_in[7];
  const float *w1c = (const float*)d_in[8],  *g1c = (const float*)d_in[9],  *b1c = (const float*)d_in[10];
  const float *w2a = (const float*)d_in[11], *g2a = (const float*)d_in[12], *b2a = (const float*)d_in[13];
  const float *w2b = (const float*)d_in[14], *g2b = (const float*)d_in[15], *b2b = (const float*)d_in[16];
  const float *w2c = (const float*)d_in[17], *g2c = (const float*)d_in[18], *b2c = (const float*)d_in[19];
  const float *wpos = (const float*)d_in[20], *bpos = (const float*)d_in[21];
  const float *wq = (const float*)d_in[22], *wk = (const float*)d_in[23];
  const float *wv = (const float*)d_in[24], *bv = (const float*)d_in[25];
  const float *wt = (const float*)d_in[26], *bt = (const float*)d_in[27];
  const float *gfe = (const float*)d_in[28], *bfe = (const float*)d_in[29];
  float* out = (float*)d_out;

  // workspace bump allocator
  char* ws = (char*)d_ws;
  size_t off = 0;
  auto alloc = [&](size_t bytes) -> void* { void* p = ws + off; off = align256(off + bytes); return p; };

  float* dist   = (float*)alloc(sizeof(float) * (size_t)B * N);
  int*   fidx1  = (int*)  alloc(sizeof(int)   * (size_t)B * S1);
  float* l1_xyz = (float*)alloc(sizeof(float) * (size_t)B * S1 * 3);
  int*   gidx1  = (int*)  alloc(sizeof(int)   * (size_t)B * S1 * K1);
  float* l1_f   = (float*)alloc(sizeof(float) * (size_t)B * S1 * C1F);
  int*   fidx2  = (int*)  alloc(sizeof(int)   * (size_t)B * S2);
  float* l2_xyz = (float*)alloc(sizeof(float) * (size_t)B * S2 * 3);
  int*   gidx2  = (int*)  alloc(sizeof(int)   * (size_t)B * S2 * K2);

  typedef _Float16 h16;
  h16* l2_fh = (h16*)alloc(sizeof(h16) * (size_t)B * S2 * C2F);
  // pre-padded f16 weights
  h16* w1a_h = (h16*)alloc(sizeof(h16) * 64 * 32);
  h16* w1b_h = (h16*)alloc(sizeof(h16) * 64 * 64);
  h16* w1c_h = (h16*)alloc(sizeof(h16) * 128 * 64);
  h16* w2a_h = (h16*)alloc(sizeof(h16) * 128 * 160);
  h16* w2b_h = (h16*)alloc(sizeof(h16) * 128 * 128);
  h16* w2c_h = (h16*)alloc(sizeof(h16) * 256 * 128);
  h16* wpos_h = (h16*)alloc(sizeof(h16) * 256 * 32);
  h16* wq_h = (h16*)alloc(sizeof(h16) * 64 * 256);
  h16* wk_h = (h16*)alloc(sizeof(h16) * 64 * 256);
  h16* wv_h = (h16*)alloc(sizeof(h16) * 256 * 256);
  h16* wt_h = (h16*)alloc(sizeof(h16) * 256 * 256);
  // f16 FEM activations
  h16* xyzp = (h16*)alloc(sizeof(h16) * (size_t)B * S2 * 32);   // l2_xyz padded 3->32
  h16* xh   = (h16*)alloc(sizeof(h16) * (size_t)B * S2 * C2F);
  h16* qh   = (h16*)alloc(sizeof(h16) * (size_t)B * S2 * 64);
  h16* kh   = (h16*)alloc(sizeof(h16) * (size_t)B * S2 * 64);
  h16* vh   = (h16*)alloc(sizeof(h16) * (size_t)B * S2 * C2F);
  h16* atth = (h16*)alloc(sizeof(h16) * (size_t)B * S2 * S2);
  h16* xrh  = (h16*)alloc(sizeof(h16) * (size_t)B * S2 * C2F);
  h16* th   = (h16*)alloc(sizeof(h16) * (size_t)B * S2 * C2F);
  h16* dh   = (h16*)alloc(sizeof(h16) * (size_t)B * S2 * C2F);

  auto cvt = [&](const float* src, h16* dst, int rows, int kact, int kpad) {
    int n = rows * kpad;
    cvt_pad_kernel<<<(n + 255) / 256, 256, 0, stream>>>(src, dst, rows, kact, kpad);
  };
  // one-time weight conversion to padded f16
  cvt(w1a, w1a_h, 64, 6, 32);
  cvt(w1b, w1b_h, 64, 64, 64);
  cvt(w1c, w1c_h, 128, 64, 64);
  cvt(w2a, w2a_h, 128, 131, 160);
  cvt(w2b, w2b_h, 128, 128, 128);
  cvt(w2c, w2c_h, 256, 128, 128);
  cvt(wpos, wpos_h, 256, 3, 32);
  cvt(wq, wq_h, 64, 256, 256);
  cvt(wk, wk_h, 64, 256, 256);
  cvt(wv, wv_h, 256, 256, 256);
  cvt(wt, wt_h, 256, 256, 256);

  // ---- SA1 ----
  fps_kernel<<<B, 256, 0, stream>>>(xyz, N, S1, dist, fidx1, l1_xyz);
  {
    int total = B * S1;
    ball_query_kernel<<<(total + 255) / 256, 256, 0, stream>>>(xyz, N, l1_xyz, S1, total,
                                                               0.2f * 0.2f, K1, gidx1);
  }
  sa_mlp_kernel<32, 6, 32, 64, 64, 128, 64, 64, float><<<B * S1, BLOCK, 0, stream>>>(
      xyz, N, xyz, 3, l1_xyz, gidx1, S1,
      w1a_h, g1a, b1a, w1b_h, g1b, b1b, w1c_h, g1c, b1c, l1_f);

  // ---- SA2 ----
  fps_kernel<<<B, 256, 0, stream>>>(l1_xyz, S1, S2, dist, fidx2, l2_xyz);
  {
    int total = B * S2;
    ball_query_kernel<<<(total + 255) / 256, 256, 0, stream>>>(l1_xyz, S1, l2_xyz, S2, total,
                                                               0.4f * 0.4f, K2, gidx2);
  }
  sa_mlp_kernel<64, 131, 160, 128, 128, 256, 160, 128, _Float16><<<B * S2, BLOCK, 0, stream>>>(
      l1_xyz, S1, l1_f, C1F, l2_xyz, gidx2, S2,
      w2a_h, g2a, b2a, w2b_h, g2b, b2b, w2c_h, g2c, b2c, l2_fh);

  // l2_xyz -> padded f16 A operand for the positional GEMM
  cvt(l2_xyz, xyzp, B * S2, 3, 32);

  // ---- FEM (offset attention), all GEMMs f16 WMMA (16x32 tile per wave) ----
  auto gtiles = [](int M, int Nn) { return ((M / 16) * (Nn / 32) + NWAVES - 1) / NWAVES; };

  // x = l2_f + l2_xyz @ wpos^T + bpos      [B,128,256]
  gemm_h_kernel<false, false><<<dim3(gtiles(S2, C2F), B), BLOCK, 0, stream>>>(
      xyzp, 32, (long)S2 * 32, wpos_h, 32, 0, xh, C2F, (long)S2 * C2F,
      S2, C2F, 32, bpos, l2_fh, (long)S2 * C2F);
  // q = x @ wq^T ; k = x @ wk^T   [B,128,64]
  gemm_h_kernel<false, false><<<dim3(gtiles(S2, 64), B), BLOCK, 0, stream>>>(
      xh, C2F, (long)S2 * C2F, wq_h, C2F, 0, qh, 64, (long)S2 * 64,
      S2, 64, C2F, nullptr, nullptr, 0);
  gemm_h_kernel<false, false><<<dim3(gtiles(S2, 64), B), BLOCK, 0, stream>>>(
      xh, C2F, (long)S2 * C2F, wk_h, C2F, 0, kh, 64, (long)S2 * 64,
      S2, 64, C2F, nullptr, nullptr, 0);
  // v = x @ wv^T + bv  [B,128,256]
  gemm_h_kernel<false, false><<<dim3(gtiles(S2, C2F), B), BLOCK, 0, stream>>>(
      xh, C2F, (long)S2 * C2F, wv_h, C2F, 0, vh, C2F, (long)S2 * C2F,
      S2, C2F, C2F, bv, nullptr, 0);
  // att = q @ k^T  [B,128,128]  (k row-major [j,64] is exactly the B operand)
  gemm_h_kernel<false, false><<<dim3(gtiles(S2, S2), B), BLOCK, 0, stream>>>(
      qh, 64, (long)S2 * 64, kh, 64, (long)S2 * 64, atth, S2, (long)S2 * S2,
      S2, S2, 64, nullptr, nullptr, 0);
  attn_softmax_kernel<<<B, S2, 0, stream>>>(atth, S2);
  // x_r[j,c] = sum_i att[i,j] * v[i,c]  -> TA (att^T), TB (v as [i,c])
  gemm_h_kernel<true, true><<<dim3(gtiles(S2, C2F), B), BLOCK, 0, stream>>>(
      atth, S2, (long)S2 * S2, vh, C2F, (long)S2 * C2F, xrh, C2F, (long)S2 * C2F,
      S2, C2F, S2, nullptr, nullptr, 0);
  // t = x - x_r
  {
    int n = B * S2 * C2F;
    sub_h_kernel<<<(n + 255) / 256, 256, 0, stream>>>(xh, xrh, th, n);
  }
  // d = t @ wt^T + bt
  gemm_h_kernel<false, false><<<dim3(gtiles(S2, C2F), B), BLOCK, 0, stream>>>(
      th, C2F, (long)S2 * C2F, wt_h, C2F, 0, dh, C2F, (long)S2 * C2F,
      S2, C2F, C2F, bt, nullptr, 0);
  // out = max_i(x + relu(d*gamma+beta)) + noise - 0.5
  {
    int n = B * C2F;
    final_kernel<<<(n + 255) / 256, 256, 0, stream>>>(xh, dh, gfe, bfe, noise, out, S2, C2F);
  }
}
